// SBMemoryRouter_28587302323142
// MI455X (gfx1250) — matrix operand order repaired and checked by reading, not compiled, not run on hardware
//
#include <hip/hip_runtime.h>
#include <cstdint>
#include <cstddef>

typedef __attribute__((ext_vector_type(8)))  float  v8f;
typedef __attribute__((ext_vector_type(8)))  __bf16 v8bf;
typedef __attribute__((ext_vector_type(16))) __bf16 v16bf;

#define BB   128      // batch
#define WW   128      // working slots
#define SS   131072   // semantic slots
#define DD   256      // dim
#define TOPK 16
#define NCOL 128          // semantic columns per block in k_semantic
#define NBLK (SS / NCOL)  // 1024

// ---- helpers -------------------------------------------------------------
static __device__ __forceinline__ unsigned bf16_rne(float f) {
    unsigned u = __float_as_uint(f);
    unsigned r = u + 0x7FFFu + ((u >> 16) & 1u);
    return r >> 16;
}
static __device__ __forceinline__ float bf16_to_f(unsigned hbits) {
    return __uint_as_float(hbits << 16);
}
// order-preserving float -> uint
static __device__ __forceinline__ unsigned sortKey(float f) {
    unsigned u = __float_as_uint(f);
    return (u & 0x80000000u) ? ~u : (u | 0x80000000u);
}
static __device__ __forceinline__ float unsortKey(unsigned k) {
    return __uint_as_float((k & 0x80000000u) ? (k & 0x7FFFFFFFu) : ~k);
}
// larger score wins; on tie, smaller index wins (matches jax top_k)
static __device__ __forceinline__ unsigned long long packKey(float s, int idx) {
    return ((unsigned long long)sortKey(s) << 32) |
           (unsigned long long)(0xFFFFFFFFu - (unsigned)idx);
}

// ---- K1: q projection + normalize + bf16 hi/lo split ---------------------
__global__ void k_qproj(const float* __restrict__ cur, const float* __restrict__ prev,
                        const float* __restrict__ Wq,
                        float* __restrict__ qn, unsigned short* __restrict__ qhi,
                        unsigned short* __restrict__ qlo) {
    __shared__ float cat[2 * DD];
    __shared__ float red[DD];
    int b = blockIdx.x, t = threadIdx.x;
    cat[t]      = cur[b * DD + t];
    cat[DD + t] = prev[b * DD + t];
    __syncthreads();
    const float* wr = Wq + (size_t)t * (2 * DD);
    float acc = 0.f;
    #pragma unroll 4
    for (int j = 0; j < 2 * DD; ++j) acc += cat[j] * wr[j];
    red[t] = acc * acc;
    __syncthreads();
    for (int s = DD / 2; s > 0; s >>= 1) { if (t < s) red[t] += red[t + s]; __syncthreads(); }
    float sc = 1.f / fmaxf(sqrtf(red[0]), 1e-6f);
    float v = acc * sc;
    qn[b * DD + t] = v;
    unsigned hi = bf16_rne(v);
    qhi[b * DD + t] = (unsigned short)hi;
    qlo[b * DD + t] = (unsigned short)bf16_rne(v - bf16_to_f(hi));
}

// ---- K2: working scores (norm folded) ------------------------------------
__global__ void k_wscore(const float* __restrict__ qn, const float* __restrict__ wkeys,
                         float* __restrict__ wsc) {
    int b = blockIdx.x;
    int wave = threadIdx.x >> 5, lane = threadIdx.x & 31;
    int n = blockIdx.y * 8 + wave;
    const float* kp = wkeys + ((size_t)b * WW + n) * DD + lane * 8;
    const float* qp = qn + (size_t)b * DD + lane * 8;
    float dot = 0.f, ss = 0.f;
    #pragma unroll
    for (int i = 0; i < 8; ++i) { float kv = kp[i]; dot += kv * qp[i]; ss += kv * kv; }
    #pragma unroll
    for (int m = 1; m <= 16; m <<= 1) { dot += __shfl_xor(dot, m, 32); ss += __shfl_xor(ss, m, 32); }
    if (lane == 0) wsc[b * WW + n] = dot / fmaxf(sqrtf(ss), 1e-6f);
}

// ---- K3: semantic GEMM (split-bf16 WMMA) + fused per-row top-16 ----------
__global__ void __launch_bounds__(256, 1)
k_semantic(const float* __restrict__ skeys,
           const unsigned short* __restrict__ qhi,
           const unsigned short* __restrict__ qlo,
           float2* __restrict__ cand) {
    __shared__ __align__(32) unsigned short Ahi[BB * DD];   // 64 KB
    __shared__ __align__(32) unsigned short Alo[BB * DD];   // 64 KB
    __shared__ __align__(32) unsigned short Bhi[NCOL * 32]; // 8 KB (K-chunk)
    __shared__ __align__(32) unsigned short Blo[NCOL * 32]; // 8 KB
    __shared__ float rsld[NCOL];
    int t = threadIdx.x;
    int wave = t >> 5, lane = t & 31;
    int g = lane >> 4, lg = lane & 15;
    int sBase = blockIdx.x * NCOL;

    // stage A (all 128 normalized q rows, hi+lo)
    {
        const uint4* sh = (const uint4*)qhi;
        const uint4* sl = (const uint4*)qlo;
        uint4* dh = (uint4*)Ahi; uint4* dl = (uint4*)Alo;
        #pragma unroll
        for (int i = 0; i < 16; ++i) { dh[t + 256 * i] = sh[t + 256 * i]; dl[t + 256 * i] = sl[t + 256 * i]; }
    }
    // per-column inverse norms (one wave handles 16 columns)
    for (int c = 0; c < 16; ++c) {
        int col = wave * 16 + c;
        const float* kp = skeys + (size_t)(sBase + col) * DD + lane * 8;
        float ss = 0.f;
        #pragma unroll
        for (int i = 0; i < 8; ++i) { float kv = kp[i]; ss += kv * kv; }
        #pragma unroll
        for (int m = 1; m <= 16; m <<= 1) ss += __shfl_xor(ss, m, 32);
        if (lane == 0) rsld[col] = 1.f / fmaxf(sqrtf(ss), 1e-6f);
    }
    __syncthreads();

    v8f acc[8];
    #pragma unroll
    for (int i = 0; i < 8; ++i) acc[i] = (v8f){0.f,0.f,0.f,0.f,0.f,0.f,0.f,0.f};

    for (int kc = 0; kc < DD / 32; ++kc) {
        int k0 = kc * 32;
        // stage B K-chunk: scale by 1/norm, split fp32 -> bf16 hi/lo
        {
            int col = t >> 1, kb = (t & 1) * 16;
            const float* src = skeys + (size_t)(sBase + col) * DD + k0 + kb;
            float scale = rsld[col];
            unsigned* ph = (unsigned*)&Bhi[col * 32 + kb];
            unsigned* pl = (unsigned*)&Blo[col * 32 + kb];
            #pragma unroll
            for (int p = 0; p < 8; ++p) {
                float f0 = src[2 * p] * scale, f1 = src[2 * p + 1] * scale;
                unsigned h0 = bf16_rne(f0), h1 = bf16_rne(f1);
                unsigned l0 = bf16_rne(f0 - bf16_to_f(h0));
                unsigned l1 = bf16_rne(f1 - bf16_to_f(h1));
                ph[p] = h0 | (h1 << 16);
                pl[p] = l0 | (l1 << 16);
            }
        }
        __syncthreads();
        // A fragment (16-bit A 16x32 layout: K{0..7,16..23} lanes 0-15, +8 lanes 16-31)
        int abase = (wave * 16 + lg) * DD + k0 + 8 * g;
        v8bf a0h = *(const v8bf*)&Ahi[abase];
        v8bf a1h = *(const v8bf*)&Ahi[abase + 16];
        v8bf a0l = *(const v8bf*)&Alo[abase];
        v8bf a1l = *(const v8bf*)&Alo[abase + 16];
        v16bf ahi, alo;
        #pragma unroll
        for (int i = 0; i < 8; ++i) { ahi[i] = a0h[i]; ahi[8 + i] = a1h[i]; alo[i] = a0l[i]; alo[8 + i] = a1l[i]; }
        #pragma unroll
        for (int t8 = 0; t8 < 8; ++t8) {
            // B 32x16 layout: lane = N, K{0..7,16..23} lanes 0-15, +8 lanes 16-31
            int off = (t8 * 16 + lg) * 32 + 8 * g;
            v8bf b0h = *(const v8bf*)&Bhi[off];
            v8bf b1h = *(const v8bf*)&Bhi[off + 16];
            v8bf b0l = *(const v8bf*)&Blo[off];
            v8bf b1l = *(const v8bf*)&Blo[off + 16];
            v16bf bhi, blo;
            #pragma unroll
            for (int i = 0; i < 8; ++i) { bhi[i] = b0h[i]; bhi[8 + i] = b1h[i]; blo[i] = b0l[i]; blo[8 + i] = b1l[i]; }
            acc[t8] = __builtin_amdgcn_wmma_f32_16x16x32_bf16(false, ahi, false, bhi, (short)0, acc[t8], false, false);
            acc[t8] = __builtin_amdgcn_wmma_f32_16x16x32_bf16(false, ahi, false, blo, (short)0, acc[t8], false, false);
            acc[t8] = __builtin_amdgcn_wmma_f32_16x16x32_bf16(false, alo, false, bhi, (short)0, acc[t8], false, false);
        }
        __syncthreads();
    }

    // per-row top-16 from accumulators (rows live entirely in one half-wave)
    #pragma unroll
    for (int j = 0; j < 8; ++j) {
        int row = wave * 16 + j + 8 * g;
        unsigned long long keys[8];
        #pragma unroll
        for (int t8 = 0; t8 < 8; ++t8) {
            int gidx = WW + sBase + t8 * 16 + lg;  // index into concat [working|semantic]
            keys[t8] = packKey(acc[t8][j], gidx);
        }
        unsigned consumed = 0;
        for (int it = 0; it < TOPK; ++it) {
            unsigned long long best = 0ull; int bt = -1;
            #pragma unroll
            for (int t8 = 0; t8 < 8; ++t8)
                if (!(consumed & (1u << t8)) && keys[t8] > best) { best = keys[t8]; bt = t8; }
            unsigned long long m = best;
            #pragma unroll
            for (int msk = 1; msk <= 8; msk <<= 1) {
                unsigned long long o = __shfl_xor(m, msk, 32);
                if (o > m) m = o;
            }
            if (bt >= 0 && m == best) consumed |= 1u << bt;
            if (lg == 0) {
                float sc = unsortKey((unsigned)(m >> 32));
                int idx = (int)(0xFFFFFFFFu - (unsigned)(m & 0xFFFFFFFFu));
                cand[((size_t)row * NBLK + blockIdx.x) * TOPK + it] = make_float2(sc, __int_as_float(idx));
            }
        }
    }
}

// ---- K4: exact merge, softmax, gather, outputs ---------------------------
__global__ void k_merge(const float* __restrict__ wsc, const float2* __restrict__ cand,
                        const float* __restrict__ wvals, const float* __restrict__ svals,
                        float* __restrict__ out, float* __restrict__ cnt) {
    __shared__ float shs[256 * TOPK];
    __shared__ int   shi[256 * TOPK];
    __shared__ float topS[TOPK]; __shared__ int topI[TOPK];
    __shared__ float wei[TOPK];  __shared__ float esum;
    int b = blockIdx.x, t = threadIdx.x;
    const int MTOT = WW + NBLK * TOPK;

    float ls[TOPK]; int li[TOPK];
    #pragma unroll
    for (int i = 0; i < TOPK; ++i) { ls[i] = -__builtin_inff(); li[i] = 0x7FFFFFFF; }
    for (int e = t; e < MTOT; e += 256) {
        float s; int idx;
        if (e < WW) { s = wsc[b * WW + e]; idx = e; }
        else { float2 c = cand[(size_t)b * (NBLK * TOPK) + (e - WW)]; s = c.x; idx = __float_as_int(c.y); }
        if (s > ls[TOPK - 1] || (s == ls[TOPK - 1] && idx < li[TOPK - 1])) {
            int pos = TOPK - 1;
            while (pos > 0 && (s > ls[pos - 1] || (s == ls[pos - 1] && idx < li[pos - 1]))) {
                ls[pos] = ls[pos - 1]; li[pos] = li[pos - 1]; --pos;
            }
            ls[pos] = s; li[pos] = idx;
        }
    }
    #pragma unroll
    for (int i = 0; i < TOPK; ++i) { shs[t * TOPK + i] = ls[i]; shi[t * TOPK + i] = li[i]; }
    __syncthreads();

    if (t < 32) {
        float fs[TOPK]; int fi[TOPK];
        #pragma unroll
        for (int i = 0; i < TOPK; ++i) { fs[i] = -__builtin_inff(); fi[i] = 0x7FFFFFFF; }
        for (int lst = t; lst < 256; lst += 32) {
            for (int i = 0; i < TOPK; ++i) {
                float s = shs[lst * TOPK + i]; int idx = shi[lst * TOPK + i];
                if (s > fs[TOPK - 1] || (s == fs[TOPK - 1] && idx < fi[TOPK - 1])) {
                    int pos = TOPK - 1;
                    while (pos > 0 && (s > fs[pos - 1] || (s == fs[pos - 1] && idx < fi[pos - 1]))) {
                        fs[pos] = fs[pos - 1]; fi[pos] = fi[pos - 1]; --pos;
                    }
                    fs[pos] = s; fi[pos] = idx;
                } else break; // source lists are sorted descending
            }
        }
        int ptr = 0;
        for (int it = 0; it < TOPK; ++it) {
            unsigned long long key = (ptr < TOPK) ? packKey(fs[ptr], fi[ptr]) : 0ull;
            unsigned long long m = key;
            #pragma unroll
            for (int msk = 1; msk <= 16; msk <<= 1) {
                unsigned long long o = __shfl_xor(m, msk, 32);
                if (o > m) m = o;
            }
            if (m == key && ptr < TOPK) ++ptr;
            if (t == 0) {
                topS[it] = unsortKey((unsigned)(m >> 32));
                topI[it] = (int)(0xFFFFFFFFu - (unsigned)(m & 0xFFFFFFFFu));
            }
        }
    }
    __syncthreads();
    if (t < TOPK) shs[t] = __expf(topS[t] - topS[0]);   // topS[0] is the max
    __syncthreads();
    if (t == 0) { float s = 0.f; for (int i = 0; i < TOPK; ++i) s += shs[i]; esum = s; }
    __syncthreads();
    if (t < TOPK) wei[t] = shs[t] / esum;
    __syncthreads();

    float mr = 0.f;
    #pragma unroll
    for (int i = 0; i < TOPK; ++i) {
        int idx = topI[i];
        float v = (idx < WW) ? wvals[((size_t)b * WW + idx) * DD + t]
                             : svals[(size_t)(idx - WW) * DD + t];
        mr += wei[i] * v;
    }
    out[(size_t)b * DD + t] = mr;
    if (t < TOPK) {
        out[BB * DD + b * TOPK + t]                 = (float)topI[t];
        out[BB * DD + BB * TOPK + b * TOPK + t]     = topS[t];
        out[BB * DD + 2 * BB * TOPK + b * TOPK + t] = wei[t];
    }
    if (t == 0) {
        int c = 0;
        for (int i = 0; i < TOPK; ++i) c += (topI[i] < WW) ? 1 : 0;
        cnt[b] = (float)c;
    }
}

// ---- K5: working ratio ----------------------------------------------------
__global__ void k_ratio(const float* __restrict__ cnt, float* __restrict__ out) {
    __shared__ float red[BB];
    int t = threadIdx.x;
    red[t] = cnt[t];
    __syncthreads();
    for (int s = BB / 2; s > 0; s >>= 1) { if (t < s) red[t] += red[t + s]; __syncthreads(); }
    if (t == 0) out[BB * DD + 3 * BB * TOPK] = red[0] / (float)(BB * TOPK);
}

// ---- launch ---------------------------------------------------------------
extern "C" void kernel_launch(void* const* d_in, const int* in_sizes, int n_in,
                              void* d_out, int out_size, void* d_ws, size_t ws_size,
                              hipStream_t stream) {
    (void)in_sizes; (void)n_in; (void)out_size; (void)ws_size;
    const float* cur  = (const float*)d_in[0];
    const float* prev = (const float*)d_in[1];
    const float* wk   = (const float*)d_in[2];
    const float* wv   = (const float*)d_in[3];
    const float* sk   = (const float*)d_in[4];
    const float* sv   = (const float*)d_in[5];
    const float* Wq   = (const float*)d_in[6];
    float* out = (float*)d_out;

    char* ws = (char*)d_ws;
    float*          qn   = (float*)(ws + 0);                 // 128*256*4   = 131072
    unsigned short* qhi  = (unsigned short*)(ws + 131072);   // 65536
    unsigned short* qlo  = (unsigned short*)(ws + 196608);   // 65536
    float*          wsc  = (float*)(ws + 262144);            // 65536
    float*          cnt  = (float*)(ws + 327680);            // 512
    float2*         cand = (float2*)(ws + 328192);           // 128*1024*16*8 = 16 MB

    k_qproj   <<<BB, DD, 0, stream>>>(cur, prev, Wq, qn, qhi, qlo);
    k_wscore  <<<dim3(BB, 16), 256, 0, stream>>>(qn, wk, wsc);
    k_semantic<<<NBLK, 256, 0, stream>>>(sk, qhi, qlo, cand);
    k_merge   <<<BB, 256, 0, stream>>>(wsc, cand, wv, sv, out, cnt);
    k_ratio   <<<1, BB, 0, stream>>>(cnt, out);
}
